// GyroNLEPredictor_20598663151936
// MI455X (gfx1250) — matrix operand order repaired and checked by reading, not compiled, not run on hardware
//
#include <hip/hip_runtime.h>
#include <math.h>

typedef __attribute__((ext_vector_type(16))) _Float16 v16h;
typedef __attribute__((ext_vector_type(8)))  _Float16 v8h;
typedef __attribute__((ext_vector_type(8)))  float    v8f;

#define TRN   8
#define HID   64
#define PHI_N 23
#define BND   5.0f
#define LN_P_OUT_C (-1.4469189829363254f)   // -log(2.5 - (-1.75))

// ---- dynamic LDS layout (bytes) ----
#define OFF_W2T   0                          // 8*64*64 f16 = 65536
#define OFF_W3T   65536                      // 8*32*64 f16 = 32768
#define OFF_W1P   98304                      // 8*64*32 f16 = 32768 (K=32 padded, k==3 carries b1)
#define OFF_B2    131072                     // 8*64 f32 = 2048
#define OFF_B3    133120                     // 8*32 f32 = 1024
#define OFF_STAGE 134144                     // per-wave 8704B: ctx@0(384) | h1@384(2048) | h2@2432(2048) | phi@4480(4096)
#define STAGE_STRIDE 8704
#define OFF_RED   (OFF_STAGE + 8*STAGE_STRIDE)   // 203776: 8 f32
#define SMEM_TOTAL (OFF_RED + 64)

// A-fragment (16x32 f16): lane m (half=0): K = s*32 + {0..7, 16..23}; half=1: +8
__device__ __forceinline__ v16h afrag_ld(const _Float16* row, int s, int half) {
  const v8h lo = *(const v8h*)(row + s*32 + half*8);
  const v8h hi = *(const v8h*)(row + s*32 + 16 + half*8);
  return __builtin_shufflevector(lo, hi, 0,1,2,3,4,5,6,7,8,9,10,11,12,13,14,15);
}
// B-fragment (32x16 f16): lane n (half=0): K = s*32 + {0..7, 16..23}; half=1: +8
__device__ __forceinline__ v16h bfrag_ld(const _Float16* row, int s, int half) {
  const _Float16* p = row + s*32 + half*8;
  const v8h lo = *(const v8h*)(p);
  const v8h hi = *(const v8h*)(p + 16);
  return __builtin_shufflevector(lo, hi, 0,1,2,3,4,5,6,7,8,9,10,11,12,13,14,15);
}
__device__ __forceinline__ v8f wmma16(v16h a, v16h b, v8f c) {
  return __builtin_amdgcn_wmma_f32_16x16x32_f16(false, a, false, b, (short)0, c, false, false);
}
__device__ __forceinline__ float relu(float x) { return (x > 0.0f) ? x : 0.0f; }

__global__ void zero_out_kernel(float* out) { out[0] = 0.0f; }

__global__ __launch_bounds__(256, 1)
void gyro_nsf_fused(const float* __restrict__ z_in,
                    const float* __restrict__ log_age,
                    const float* __restrict__ bprp0,
                    const float* __restrict__ log_err,
                    const float* __restrict__ mem_prob,
                    const float* __restrict__ W1,
                    const float* __restrict__ b1,
                    const float* __restrict__ W2,
                    const float* __restrict__ b2,
                    const float* __restrict__ W3,
                    const float* __restrict__ b3,
                    float* __restrict__ out,
                    int B)
{
  extern __shared__ char smem[];
  _Float16* W2T = (_Float16*)(smem + OFF_W2T);
  _Float16* W3T = (_Float16*)(smem + OFF_W3T);
  _Float16* W1P = (_Float16*)(smem + OFF_W1P);
  float* b2s = (float*)(smem + OFF_B2);
  float* b3s = (float*)(smem + OFF_B3);
  float* redbuf = (float*)(smem + OFF_RED);

  const int tid  = threadIdx.x;
  const int lane = tid & 31;
  const int wid  = tid >> 5;
  const int half = lane >> 4;
  const int mlo  = lane & 15;

  char* stage = smem + OFF_STAGE + wid * STAGE_STRIDE;
  float*     ctxs = (float*)(stage);
  _Float16*  h1s  = (_Float16*)(stage + 384);
  _Float16*  h2s  = (_Float16*)(stage + 2432);
  float*     phis = (float*)(stage + 4480);

  // ---- cooperative weight staging: fp32 -> fp16, transposed to [t][n][k] ----
  for (int i = tid; i < TRN*HID*HID; i += blockDim.x) {
    int t = i >> 12, r = i & 4095, n = r >> 6, k = r & 63;
    W2T[i] = (_Float16)W2[(t << 12) + (k << 6) + n];
  }
  for (int i = tid; i < TRN*32*HID; i += blockDim.x) {
    int t = i >> 11, r = i & 2047, n = r >> 6, k = r & 63;
    W3T[i] = (n < PHI_N) ? (_Float16)W3[(t*HID + k)*PHI_N + n] : (_Float16)0.0f;
  }
  for (int i = tid; i < TRN*HID*32; i += blockDim.x) {
    int t = i >> 11, r = i & 2047, n = r >> 5, k = r & 31;
    float v = (k < 3) ? W1[(t*3 + k)*HID + n] : ((k == 3) ? b1[t*HID + n] : 0.0f);
    W1P[i] = (_Float16)v;                       // bias folded in as K-row 3
  }
  for (int i = tid; i < TRN*HID; i += blockDim.x) b2s[i] = b2[i];
  for (int i = tid; i < TRN*32;  i += blockDim.x) {
    int t = i >> 5, n = i & 31;
    b3s[i] = (n < PHI_N) ? b3[t*PHI_N + n] : 0.0f;
  }
  __syncthreads();

  const int wavesPerBlk = blockDim.x >> 5;
  const int gw     = blockIdx.x * wavesPerBlk + wid;
  const int stride = gridDim.x * blockDim.x;          // samples per pass
  const int niter  = (B + stride - 1) / stride;       // uniform across block

  float accsum = 0.0f;
  const _Float16 hz = (_Float16)0.0f;
  const v8f vz = {};

  for (int it = 0; it < niter; ++it) {
    __syncthreads();
    const int  sidx  = gw*32 + it*stride + lane;
    const int  cidx  = (sidx < B) ? sidx : (B - 1);
    const bool valid = sidx < B;
    float zv = z_in[cidx];
    float pm = mem_prob[cidx];
    ctxs[lane*3 + 0] = log_age[cidx];
    ctxs[lane*3 + 1] = bprp0[cidx];
    ctxs[lane*3 + 2] = log_err[cidx];
    __syncthreads();

    // ---- build ctx A-fragments once per iteration (K=32: c0,c1,c2,1,0...0) ----
    v16h actx[2];
    #pragma unroll
    for (int u = 0; u < 2; ++u) {
      const float c0 = ctxs[(u*16 + mlo)*3 + 0];
      const float c1 = ctxs[(u*16 + mlo)*3 + 1];
      const float c2 = ctxs[(u*16 + mlo)*3 + 2];
      const bool lo = (half == 0);                 // K 0..7 live in half==0 lanes
      #pragma unroll
      for (int p = 0; p < 16; ++p) actx[u][p] = hz;
      actx[u][0] = lo ? (_Float16)c0 : hz;
      actx[u][1] = lo ? (_Float16)c1 : hz;
      actx[u][2] = lo ? (_Float16)c2 : hz;
      actx[u][3] = lo ? (_Float16)1.0f : hz;       // bias row
    }

    float logdet = 0.0f;
    for (int t = 0; t < TRN; ++t) {
      #pragma unroll
      for (int u = 0; u < 2; ++u) {                 // two M=16 sample tiles
        // ---- layer 1: (16x32)@(32x64) — 4 independent WMMAs, then stage ----
        v8f acc1[4];
        #pragma unroll
        for (int j = 0; j < 4; ++j)
          acc1[j] = wmma16(actx[u], bfrag_ld(&W1P[(t*HID + j*16 + mlo)*32], 0, half), vz);
        #pragma unroll
        for (int j = 0; j < 4; ++j) {
          #pragma unroll
          for (int r = 0; r < 8; ++r)
            h1s[(r + 8*half)*HID + j*16 + mlo] = (_Float16)relu(acc1[j][r]);
        }
        __syncthreads();
        v16h af0 = afrag_ld(&h1s[mlo*HID], 0, half);
        v16h af1 = afrag_ld(&h1s[mlo*HID], 1, half);
        // ---- layer 2: (16x64)@(64x64) — 4 independent 2-WMMA chains ----
        v8f acc2[4];
        #pragma unroll
        for (int j = 0; j < 4; ++j)
          acc2[j] = wmma16(af0, bfrag_ld(&W2T[(t*HID + j*16 + mlo)*HID], 0, half), vz);
        #pragma unroll
        for (int j = 0; j < 4; ++j)
          acc2[j] = wmma16(af1, bfrag_ld(&W2T[(t*HID + j*16 + mlo)*HID], 1, half), acc2[j]);
        #pragma unroll
        for (int j = 0; j < 4; ++j) {
          const float b2v = b2s[t*HID + j*16 + mlo];
          #pragma unroll
          for (int r = 0; r < 8; ++r)
            h2s[(r + 8*half)*HID + j*16 + mlo] = (_Float16)relu(acc2[j][r] + b2v);
        }
        __syncthreads();
        // ---- layer 3: (16x64)@(64x32) — 2 independent 2-WMMA chains ----
        v16h a30 = afrag_ld(&h2s[mlo*HID], 0, half);
        v16h a31 = afrag_ld(&h2s[mlo*HID], 1, half);
        v8f acc3[2];
        #pragma unroll
        for (int j = 0; j < 2; ++j)
          acc3[j] = wmma16(a30, bfrag_ld(&W3T[(t*32 + j*16 + mlo)*HID], 0, half), vz);
        #pragma unroll
        for (int j = 0; j < 2; ++j)
          acc3[j] = wmma16(a31, bfrag_ld(&W3T[(t*32 + j*16 + mlo)*HID], 1, half), acc3[j]);
        #pragma unroll
        for (int j = 0; j < 2; ++j) {
          const float b3v = b3s[t*32 + j*16 + mlo];
          #pragma unroll
          for (int r = 0; r < 8; ++r)
            phis[(u*16 + r + 8*half)*32 + j*16 + mlo] = acc3[j][r] + b3v;
        }
        __syncthreads();
      } // u

      // ---- RQS monotonic rational-quadratic spline: one sample per lane ----
      const float* ph = &phis[lane*32];
      float wr[8], hr[8];
      #pragma unroll
      for (int i = 0; i < 8; ++i) { wr[i] = ph[i]; hr[i] = ph[8 + i]; }
      float mw = wr[0], mh = hr[0];
      #pragma unroll
      for (int i = 1; i < 8; ++i) { mw = fmaxf(mw, wr[i]); mh = fmaxf(mh, hr[i]); }
      float sw = 0.0f, sh = 0.0f;
      #pragma unroll
      for (int i = 0; i < 8; ++i) {
        wr[i] = __expf(wr[i] - mw); sw += wr[i];
        hr[i] = __expf(hr[i] - mh); sh += hr[i];
      }
      const float iw = 1.0f / sw, ih = 1.0f / sh;
      float xk[9], yk[9], dd[9];
      xk[0] = -BND; yk[0] = -BND;
      float cw = 0.0f, chh = 0.0f;
      #pragma unroll
      for (int i = 0; i < 8; ++i) {
        cw  += wr[i]; xk[i+1] = BND * (2.0f*cw*iw  - 1.0f);
        chh += hr[i]; yk[i+1] = BND * (2.0f*chh*ih - 1.0f);
      }
      dd[0] = 1.0f; dd[8] = 1.0f;
      #pragma unroll
      for (int i = 1; i < 8; ++i) {
        const float xx = ph[15 + i];                       // derivs: cols 16..22
        dd[i] = (xx > 15.0f) ? xx : log1pf(__expf(xx));    // softplus
      }
      const bool  inside = fabsf(zv) < BND;
      const float xc = fminf(fmaxf(zv, -BND), BND);
      int kk = -1;
      #pragma unroll
      for (int i = 0; i < 8; ++i) kk += (xc >= xk[i]) ? 1 : 0;
      kk = (kk < 0) ? 0 : ((kk > 7) ? 7 : kk);
      float x0 = xk[0], x1 = xk[1], y0 = yk[0], y1 = yk[1], d0 = dd[0], d1 = dd[1];
      #pragma unroll
      for (int i = 1; i < 8; ++i) {
        const bool c = (kk == i);
        x0 = c ? xk[i]   : x0;  x1 = c ? xk[i+1] : x1;
        y0 = c ? yk[i]   : y0;  y1 = c ? yk[i+1] : y1;
        d0 = c ? dd[i]   : d0;  d1 = c ? dd[i+1] : d1;
      }
      const float wk  = x1 - x0, hk = y1 - y0;
      const float sk  = hk / wk;
      const float xi  = (xc - x0) / wk, xi1 = 1.0f - xi;
      const float den = sk + (d1 + d0 - 2.0f*sk) * xi * xi1;
      const float yv  = y0 + hk * (sk*xi*xi + d0*xi*xi1) / den;
      const float ldv = 2.0f*__logf(sk)
                      + __logf(d1*xi*xi + 2.0f*sk*xi*xi1 + d0*xi1*xi1)
                      - 2.0f*__logf(den);
      zv     = inside ? yv : zv;
      logdet += inside ? ldv : 0.0f;
    } // t

    // ---- mixture likelihood ----
    const float logp = -0.5f*zv*zv - 0.91893853320467274f + logdet;
    const float pme  = (pm == pm) ? pm : 0.9f;          // NaN -> default
    const float nfw  = pme * 0.95f;                     // p_mem * (1 - P_OUTLIER)
    const float la   = __logf(nfw) + logp;
    const float lb   = __logf(1.0f - nfw) + LN_P_OUT_C;
    const float mx   = fmaxf(la, lb);
    const float lc   = mx + log1pf(__expf(-fabsf(la - lb)));
    accsum += valid ? lc : 0.0f;
  } // it

  // ---- reduction: wave -> block -> global atomic ----
  #pragma unroll
  for (int off = 16; off > 0; off >>= 1) accsum += __shfl_down(accsum, off);
  if (lane == 0) redbuf[wid] = accsum;
  __syncthreads();
  if (tid == 0) {
    float s = 0.0f;
    for (int i = 0; i < wavesPerBlk; ++i) s += redbuf[i];
    atomicAdd(out, -s / (float)B);
  }
}

extern "C" void kernel_launch(void* const* d_in, const int* in_sizes, int n_in,
                              void* d_out, int out_size, void* d_ws, size_t ws_size,
                              hipStream_t stream) {
  (void)n_in; (void)out_size; (void)d_ws; (void)ws_size;
  const float* z   = (const float*)d_in[0];
  const float* la  = (const float*)d_in[1];
  const float* bp  = (const float*)d_in[2];
  const float* le  = (const float*)d_in[3];
  const float* mp  = (const float*)d_in[4];
  const float* W1  = (const float*)d_in[5];
  const float* b1  = (const float*)d_in[6];
  const float* W2  = (const float*)d_in[7];
  const float* b2  = (const float*)d_in[8];
  const float* W3  = (const float*)d_in[9];
  const float* b3  = (const float*)d_in[10];
  const int B = in_sizes[1];

  zero_out_kernel<<<1, 1, 0, stream>>>((float*)d_out);
  gyro_nsf_fused<<<dim3(512), dim3(256), SMEM_TOTAL, stream>>>(
      z, la, bp, le, mp, W1, b1, W2, b2, W3, b3, (float*)d_out, B);
}
